// NPG_Model_7301444403264
// MI455X (gfx1250) — matrix operand order repaired and checked by reading, not compiled, not run on hardware
//
#include <hip/hip_runtime.h>

typedef _Float16 v16h __attribute__((ext_vector_type(16)));
typedef _Float16 v8h  __attribute__((ext_vector_type(8)));
typedef float    v8f  __attribute__((ext_vector_type(8)));

#define THREADS 256
#define MT      128     // rows per block tile (one 16-row strip per wave)
#define LDH     520     // LDS activation row stride (halfs), 16B-mult, bank-skewed
#define SLDS    40      // slab row stride (halfs): 20 dwords -> conflict-free b128
#define SLABSZ  (128*SLDS)  // halfs per slab buffer
#define HID     512
#define NB      16384   // B
#define NE      4       // ensemble

// ---------------------------------------------------------------------------
// Shared WMMA layer: hout[MT x 512] = relu(hin[MT x K] @ W[K x 512] + bias)
// W transposed in global as wT[512][K] f16 (n-major) -> B fragments are two
// contiguous b128 loads. Weight slabs (128 cols x 32 K) are double-buffered in
// LDS with register-staged global prefetch: one barrier per 32-K step, global
// loads overlap the 64 WMMAs each slab feeds. Layouts per CDNA5 ISA 7.12.2.
// ---------------------------------------------------------------------------
__device__ __forceinline__ void wmma_layer(
    const _Float16* hin, int ldin, int K,
    const _Float16* __restrict__ wT,   // global [512][K] f16
    const float*    bias,              // LDS [512] f32
    _Float16*       hout, int ldout,
    _Float16*       slab)              // LDS [2*SLABSZ] f16
{
  const int tid   = threadIdx.x;
  const int lane  = tid & 31;
  const int strip = tid >> 5;    // wave = 16-row strip (MT=128 -> 8 strips)
  const int r     = lane & 15;
  const int hi    = lane >> 4;
  const int nn    = tid >> 1;    // coop slab-load role: row (n)
  const int part  = tid & 1;     // coop slab-load role: 16-half chunk

  for (int sb = 0; sb < 4; ++sb) {          // 4 slab blocks of 128 cols
    v8f acc[8];
    #pragma unroll
    for (int t = 0; t < 8; ++t)
      #pragma unroll
      for (int i = 0; i < 8; ++i) acc[t][i] = 0.f;

    const _Float16* wrow = wT + (size_t)(sb*128 + nn)*K + part*16;

    // prologue: fetch first chunk, publish into buffer 0
    v8h g0 = ((const v8h*)wrow)[0];
    v8h g1 = ((const v8h*)wrow)[1];
    __syncthreads();                        // prior users of slab done
    {
      v8h* d = (v8h*)(slab + nn*SLDS + part*16);
      d[0] = g0; d[1] = g1;
    }

    int p = 0;
    for (int k0 = 0; k0 < K; k0 += 32) {
      const bool more = (k0 + 32 < K);      // uniform across block
      if (more) {                            // prefetch next chunk to regs
        const v8h* s = (const v8h*)(wrow + (k0 + 32));
        g0 = s[0]; g1 = s[1];
      }
      __syncthreads();                      // slab[p] visible; slab[p^1] free

      const _Float16* sl = slab + p*SLABSZ;

      // A fragment 16x32 f16 (lanes 0-15: K-halves {0..7,16..23}; 16-31: +8)
      const _Float16* arow = hin + (size_t)(strip*16 + r)*ldin + k0 + hi*8;
      v8h alo = *(const v8h*)arow;
      v8h ahi = *(const v8h*)(arow + 16);
      v16h a;
      #pragma unroll
      for (int i = 0; i < 8; ++i) { a[i] = alo[i]; a[8+i] = ahi[i]; }

      #pragma unroll
      for (int t = 0; t < 8; ++t) {
        // B fragment 32x16: lane = column, K {0..7,16..23} / +8 by lane half
        const _Float16* bp = sl + (t*16 + r)*SLDS + hi*8;
        v8h blo = *(const v8h*)bp;
        v8h bhi = *(const v8h*)(bp + 16);
        v16h bf;
        #pragma unroll
        for (int i = 0; i < 8; ++i) { bf[i] = blo[i]; bf[8+i] = bhi[i]; }
        acc[t] = __builtin_amdgcn_wmma_f32_16x16x32_f16(
            false, a, false, bf, (short)0, acc[t], false, false);
      }

      if (more) {                            // publish next chunk
        v8h* d = (v8h*)(slab + (p^1)*SLABSZ + nn*SLDS + part*16);
        d[0] = g0; d[1] = g1;
        p ^= 1;
      }
    }

    // epilogue: bias + relu -> f16 LDS. D row = reg + 8*(lane>=16), col = lane&15
    #pragma unroll
    for (int t = 0; t < 8; ++t) {
      const int col = sb*128 + t*16 + r;
      const float bb = bias[col];
      #pragma unroll
      for (int i = 0; i < 8; ++i) {
        const int row = strip*16 + hi*8 + i;
        hout[(size_t)row*ldout + col] = (_Float16)fmaxf(acc[t][i] + bb, 0.f);
      }
    }
  }
}

// ---------------------------------------------------------------------------
// Prep: w1 [E][512][512] f32 (k-major) -> w1T [E][512][512] f16 (n-major)
// ---------------------------------------------------------------------------
__global__ __launch_bounds__(256) void k_transpose_w1(
    const float* __restrict__ src, _Float16* __restrict__ dst)
{
  __shared__ _Float16 tile[64*65];
  const int e  = blockIdx.z;
  const int kb = blockIdx.y * 64;
  const int nb = blockIdx.x * 64;
  const float* s = src + ((size_t)e*512 + kb)*512 + nb;
  for (int i = threadIdx.x; i < 64*64; i += 256) {
    const int rr = i >> 6, cc = i & 63;            // rr: k, cc: n
    tile[cc*65 + rr] = (_Float16)s[rr*512 + cc];
  }
  __syncthreads();
  _Float16* d = dst + ((size_t)e*512 + nb)*512 + kb;
  for (int i = threadIdx.x; i < 64*64; i += 256) {
    const int rr = i >> 6, cc = i & 63;            // rr: n, cc: k
    d[(size_t)rr*512 + cc] = tile[rr*65 + cc];
  }
}

// Prep: w0 [E][NF][512] f32 -> w0T [E][512][32] f16, K zero-padded to 32
__global__ __launch_bounds__(256) void k_prep_w0T(
    const float* __restrict__ src, _Float16* __restrict__ dst, int NF)
{
  const int idx = blockIdx.x*256 + threadIdx.x;    // E*512*32 = 65536
  if (idx >= NE*512*32) return;
  const int k = idx & 31;
  const int n = (idx >> 5) & 511;
  const int e = idx >> 14;
  dst[idx] = (k < NF) ? (_Float16)src[((size_t)e*NF + k)*512 + n] : (_Float16)0.f;
}

// ---------------------------------------------------------------------------
// Leg MLP, fused: gather -> L0(WMMA,K=32) -> L1(WMMA,K=512) -> L2(vector dot)
// ---------------------------------------------------------------------------
__global__ __launch_bounds__(256) void k_leg(
    const float* __restrict__ state, const float* __restrict__ act,
    const _Float16* __restrict__ w0T, const float* __restrict__ b0,
    const _Float16* __restrict__ w1T, const float* __restrict__ b1,
    const float* __restrict__ w2, const float* __restrict__ b2,
    const float* __restrict__ mu, const float* __restrict__ sigma,
    const float* __restrict__ mu_t, const float* __restrict__ sig_t,
    float* __restrict__ legs_diff)
{
  __shared__ __align__(16) _Float16 sh_h0[MT*LDH];
  __shared__ __align__(16) _Float16 sh_h1[MT*LDH];
  __shared__ __align__(16) _Float16 sh_x [MT*32];
  __shared__ __align__(16) _Float16 sh_sl[2*SLABSZ];
  __shared__ float sh_b0[HID];
  __shared__ float sh_b1[HID];

  const int tid  = threadIdx.x;
  const int e    = blockIdx.y;
  const int row0 = blockIdx.x * MT;

  for (int c = tid; c < HID; c += THREADS) {
    sh_b0[c] = b0[e*HID + c];
    sh_b1[c] = b1[e*HID + c];
  }
  for (int r = tid; r < MT; r += THREADS) {
    const int grow = row0 + r;
    const int ib = grow >> 14;
    const int b  = grow & (NB-1);
    const float* st = state + ((size_t)e*NB + b)*29;
    const float* ac = act   + ((size_t)e*NB + b)*8;
    float f[6];
    f[0] = st[13 + 2*ib]; f[1] = st[21 + 2*ib];
    f[2] = st[14 + 2*ib]; f[3] = st[22 + 2*ib];
    f[4] = ac[2*ib];      f[5] = ac[2*ib + 1];
    _Float16* x = sh_x + r*32;
    #pragma unroll
    for (int j = 0; j < 6; ++j)
      x[j] = (_Float16)((f[j] - mu[j]) / (sigma[j] + 1e-8f));
    #pragma unroll
    for (int j = 6; j < 32; ++j) x[j] = (_Float16)0.f;
  }
  // barriers inside wmma_layer order the LDS producer/consumer phases
  wmma_layer(sh_x,  32,  32,  w0T + (size_t)e*HID*32,  sh_b0, sh_h0, LDH, sh_sl);
  wmma_layer(sh_h0, LDH, HID, w1T + (size_t)e*HID*HID, sh_b1, sh_h1, LDH, sh_sl);
  __syncthreads();

  // stage the 4 used w2 columns into the (now free) slab as f16 [j][k]
  for (int i = tid; i < HID*4; i += THREADS) {
    const int k = i >> 2, j = i & 3;
    sh_sl[j*HID + k] = (_Float16)w2[((size_t)e*HID + k)*8 + j];
  }
  __syncthreads();

  // layer 2: only outputs 0..3 are used downstream; 8-wide dual-LDS dot
  for (int idx = tid; idx < MT*4; idx += THREADS) {
    const int r = idx >> 2, j = idx & 3;
    const _Float16* hr = sh_h1 + (size_t)r*LDH;
    const _Float16* wj = sh_sl + j*HID;
    float acc = 0.f;
    #pragma unroll 4
    for (int k0 = 0; k0 < HID; k0 += 8) {
      const v8h h = *(const v8h*)(hr + k0);
      const v8h w = *(const v8h*)(wj + k0);
      #pragma unroll
      for (int u = 0; u < 8; ++u) acc += (float)h[u] * (float)w[u];
    }
    acc += b2[e*8 + j];
    const float d = acc * (sig_t[j] + 1e-8f) + mu_t[j];
    legs_diff[((size_t)e*4*NB + row0 + r)*4 + j] = d;
  }
}

// ---------------------------------------------------------------------------
// Pose MLP, fused. Input = [state[:13], leg_format(legs_diff)[13:29]]
// ---------------------------------------------------------------------------
__global__ __launch_bounds__(256) void k_pose(
    const float* __restrict__ state,
    const float* __restrict__ legs_diff,
    const _Float16* __restrict__ w0T, const float* __restrict__ b0,
    const _Float16* __restrict__ w1T, const float* __restrict__ b1,
    const float* __restrict__ w2, const float* __restrict__ b2,
    const float* __restrict__ mu, const float* __restrict__ sigma,
    const float* __restrict__ mu_t, const float* __restrict__ sig_t,
    float* __restrict__ pose_out)
{
  __shared__ __align__(16) _Float16 sh_h0[MT*LDH];
  __shared__ __align__(16) _Float16 sh_h1[MT*LDH];
  __shared__ __align__(16) _Float16 sh_x [MT*32];
  __shared__ __align__(16) _Float16 sh_sl[2*SLABSZ];
  __shared__ float sh_b0[HID];
  __shared__ float sh_b1[HID];

  const int tid  = threadIdx.x;
  const int e    = blockIdx.y;
  const int row0 = blockIdx.x * MT;

  for (int c = tid; c < HID; c += THREADS) {
    sh_b0[c] = b0[e*HID + c];
    sh_b1[c] = b1[e*HID + c];
  }
  for (int r = tid; r < MT; r += THREADS) {
    const int b = row0 + r;
    const float* st = state + ((size_t)e*NB + b)*29;
    _Float16* x = sh_x + r*32;
    #pragma unroll
    for (int j = 0; j < 13; ++j)
      x[j] = (_Float16)((st[j] - mu[j]) / (sigma[j] + 1e-8f));
    #pragma unroll
    for (int jj = 0; jj < 16; ++jj) {
      int ib, dj;
      if (jj < 8) { ib = jj >> 1;            dj = (jj & 1) ? 2 : 0; }
      else        { const int t2 = jj - 8; ib = t2 >> 1; dj = (t2 & 1) ? 3 : 1; }
      const float v = legs_diff[(((size_t)e*4 + ib)*NB + b)*4 + dj];
      const int j = 13 + jj;
      x[j] = (_Float16)((v - mu[j]) / (sigma[j] + 1e-8f));
    }
    x[29] = (_Float16)0.f; x[30] = (_Float16)0.f; x[31] = (_Float16)0.f;
  }
  wmma_layer(sh_x,  32,  32,  w0T + (size_t)e*HID*32,  sh_b0, sh_h0, LDH, sh_sl);
  wmma_layer(sh_h0, LDH, HID, w1T + (size_t)e*HID*HID, sh_b1, sh_h1, LDH, sh_sl);
  __syncthreads();

  // stage the 13 used w2 columns into the slab as f16 [j][k]
  for (int i = tid; i < HID*13; i += THREADS) {
    const int j = i % 13, k = i / 13;
    sh_sl[j*HID + k] = (_Float16)w2[((size_t)e*HID + k)*26 + j];
  }
  __syncthreads();

  // layer 2: outputs 0..12; add state residual + de-normalize
  for (int idx = tid; idx < MT*13; idx += THREADS) {
    const int r = idx / 13, j = idx - r*13;
    const _Float16* hr = sh_h1 + (size_t)r*LDH;
    const _Float16* wj = sh_sl + j*HID;
    float acc = 0.f;
    #pragma unroll 4
    for (int k0 = 0; k0 < HID; k0 += 8) {
      const v8h h = *(const v8h*)(hr + k0);
      const v8h w = *(const v8h*)(wj + k0);
      #pragma unroll
      for (int u = 0; u < 8; ++u) acc += (float)h[u] * (float)w[u];
    }
    acc += b2[e*26 + j];
    const int b = row0 + r;
    const float p = acc * (sig_t[j] + 1e-8f) + mu_t[j]
                  + state[((size_t)e*NB + b)*29 + j];
    pose_out[((size_t)e*NB + b)*13 + j] = p;
  }
}

// ---------------------------------------------------------------------------
// Finalize: out[b,c] = mean_e next_state[e,b,c]
// cols 13..28 come from state + legs_diff via the _leg_format permutation
// ---------------------------------------------------------------------------
__global__ __launch_bounds__(256) void k_final(
    const float* __restrict__ state,
    const float* __restrict__ legs_diff,
    const float* __restrict__ pose_out,
    float* __restrict__ out)
{
  const int idx = blockIdx.x*256 + threadIdx.x;
  if (idx >= NB*29) return;
  const int b = idx / 29, c = idx - b*29;
  float s = 0.f;
  if (c < 13) {
    #pragma unroll
    for (int e = 0; e < NE; ++e)
      s += pose_out[((size_t)e*NB + b)*13 + c];
  } else {
    const int k = c - 13;
    int ib, j;
    if (k < 8) { ib = k >> 1;           j = (k & 1) ? 2 : 0; }
    else       { const int t = k - 8; ib = t >> 1; j = (t & 1) ? 3 : 1; }
    #pragma unroll
    for (int e = 0; e < NE; ++e)
      s += state[((size_t)e*NB + b)*29 + c]
         + legs_diff[(((size_t)e*4 + ib)*NB + b)*4 + j];
  }
  out[idx] = s * 0.25f;
}

// ---------------------------------------------------------------------------
extern "C" void kernel_launch(void* const* d_in, const int* in_sizes, int n_in,
                              void* d_out, int out_size, void* d_ws, size_t ws_size,
                              hipStream_t stream) {
  const float* state       = (const float*)d_in[0];
  const float* act         = (const float*)d_in[1];
  const float* wl0         = (const float*)d_in[2];
  const float* bl0         = (const float*)d_in[3];
  const float* wl1         = (const float*)d_in[4];
  const float* bl1         = (const float*)d_in[5];
  const float* wl2         = (const float*)d_in[6];
  const float* bl2         = (const float*)d_in[7];
  const float* wp0         = (const float*)d_in[8];
  const float* bp0         = (const float*)d_in[9];
  const float* wp1         = (const float*)d_in[10];
  const float* bp1         = (const float*)d_in[11];
  const float* wp2         = (const float*)d_in[12];
  const float* bp2         = (const float*)d_in[13];
  const float* mu_leg      = (const float*)d_in[14];
  const float* sigma_leg   = (const float*)d_in[15];
  const float* mu_pose     = (const float*)d_in[16];
  const float* sigma_pose  = (const float*)d_in[17];
  const float* mu_t_leg    = (const float*)d_in[18];
  const float* sigma_t_leg = (const float*)d_in[19];
  const float* mu_t_pose   = (const float*)d_in[20];
  const float* sigma_t_pose= (const float*)d_in[21];
  float* out = (float*)d_out;

  // workspace layout (bytes)
  char* ws = (char*)d_ws;
  _Float16* w1T_leg  = (_Float16*)(ws + 0);          // 4*512*512 f16 = 2 MB
  _Float16* w1T_pose = (_Float16*)(ws + 2097152);    // 2 MB
  _Float16* w0T_leg  = (_Float16*)(ws + 4194304);    // 4*512*32 f16 = 128 KB
  _Float16* w0T_pose = (_Float16*)(ws + 4325376);    // 128 KB
  float*    legs_diff= (float*)   (ws + 4456448);    // 4*65536*4 f32 = 4 MB
  float*    pose_out = (float*)   (ws + 8650752);    // 4*16384*13 f32 ~ 3.25 MB
  (void)in_sizes; (void)n_in; (void)out_size; (void)ws_size;

  k_transpose_w1<<<dim3(8,8,NE), dim3(256), 0, stream>>>(wl1, w1T_leg);
  k_transpose_w1<<<dim3(8,8,NE), dim3(256), 0, stream>>>(wp1, w1T_pose);
  k_prep_w0T<<<dim3(256), dim3(256), 0, stream>>>(wl0, w0T_leg, 6);
  k_prep_w0T<<<dim3(256), dim3(256), 0, stream>>>(wp0, w0T_pose, 29);

  k_leg<<<dim3(4*NB/MT, NE), dim3(256), 0, stream>>>(
      state, act, w0T_leg, bl0, w1T_leg, bl1, wl2, bl2,
      mu_leg, sigma_leg, mu_t_leg, sigma_t_leg, legs_diff);

  k_pose<<<dim3(NB/MT, NE), dim3(256), 0, stream>>>(
      state, legs_diff, w0T_pose, bp0, w1T_pose, bp1, wp2, bp2,
      mu_pose, sigma_pose, mu_t_pose, sigma_t_pose, pose_out);

  k_final<<<dim3((NB*29 + 255)/256), dim3(256), 0, stream>>>(
      state, legs_diff, pose_out, out);
}
